// StaticDotAttention_21131239097075
// MI455X (gfx1250) — matrix operand order repaired and checked by reading, not compiled, not run on hardware
//
#include <hip/hip_runtime.h>

// ---------------------------------------------------------------------------
// StaticDotAttention for MI455X (gfx1250), wave32, WMMA bf16 path.
// out[b,q,:] = concat( softmax(mask(relu(inp@Wi) @ relu(mem@Wm)^T / sqrt(512))) @ mem,
//                      inp[b,q,:] )
// B=32, Sq=Sk=1024, Din=Dmem=ATT=512.
// ---------------------------------------------------------------------------

typedef __attribute__((ext_vector_type(16))) __bf16 v16bf;
typedef __attribute__((ext_vector_type(8)))  float  v8f;
typedef int v4i_b128 __attribute__((vector_size(4 * sizeof(int))));

#define BATCH 32
#define SEQ   1024
#define DIM   512

// ---- gfx1250 async global->LDS staging (ASYNCcnt-tracked DMA) --------------
#if defined(__gfx1250__) && __has_builtin(__builtin_amdgcn_global_load_async_to_lds_b128)
#define USE_ASYNC_LDS 1
#endif

__device__ __forceinline__ void copy16B(unsigned short* ldst, const unsigned short* gsrc) {
#ifdef USE_ASYNC_LDS
  __builtin_amdgcn_global_load_async_to_lds_b128(
      (__attribute__((address_space(1))) v4i_b128*)(v4i_b128*)(void*)(const void*)gsrc,
      (__attribute__((address_space(3))) v4i_b128*)(v4i_b128*)(void*)ldst, 0, 0);
#else
  *(uint4*)ldst = *(const uint4*)gsrc;
#endif
}

// wait until at most the newest 16 async issues are still outstanding
__device__ __forceinline__ void async_wait_le16() {
#ifdef USE_ASYNC_LDS
  asm volatile("s_wait_asynccnt 0x10" ::: "memory");
#endif
}
__device__ __forceinline__ void async_wait_0() {
#ifdef USE_ASYNC_LDS
  asm volatile("s_wait_asynccnt 0x0" ::: "memory");
#endif
}

__device__ __forceinline__ unsigned short f32_to_bf16(float f) {
  unsigned int u = __float_as_uint(f);
  u = (u + 0x7FFFu + ((u >> 16) & 1u)) >> 16;   // round-to-nearest-even
  return (unsigned short)u;
}

__device__ __forceinline__ v8f wmma_bf16(v16bf a, v16bf b, v8f c) {
  return __builtin_amdgcn_wmma_f32_16x16x32_bf16(false, a, false, b, (short)0, c,
                                                 false, false);
}

// A-fragment (16x32 bf16, MxK): lane = half*16 + m.  Elements e<8 -> K = k0+half*8+e,
// e>=8 -> K = k0+16+half*8+(e-8).  Storage: row-major, K contiguous.
__device__ __forceinline__ v16bf load_frag_A(const unsigned short* base, int stride,
                                             int row, int k0, int half) {
  union { v16bf v; uint4 q[2]; } u;
  const char* p = (const char*)(base + row * stride + k0 + half * 8);
  u.q[0] = *(const uint4*)(p);
  u.q[1] = *(const uint4*)(p + 32);
  return u.v;
}

// B-fragment (32x16 bf16, KxN): lane = half*16 + n. Elements e<8 -> K = k0+half*8+e,
// e>=8 -> K = k0+16+half*8+(e-8).
// Storage: N-major rows (row index = n), K contiguous within a row.
__device__ __forceinline__ v16bf load_frag_B(const unsigned short* base, int stride,
                                             int nrow, int k0, int half) {
  union { v16bf v; uint4 q[2]; } u;
  const char* p = (const char*)(base + nrow * stride + k0 + half * 8);
  u.q[0] = *(const uint4*)(p);
  u.q[1] = *(const uint4*)(p + 32);
  return u.v;
}

// ---------------------------------------------------------------------------
// Kernel 1: Y[M,512](bf16) = relu( X[M,512](f32) @ W[512,512](f32) )
// Block: 256 threads (8 waves). Tile: BM=128, BN=64, BK=32.
// ---------------------------------------------------------------------------
__global__ void __launch_bounds__(256)
proj_relu_bf16(const float* __restrict__ X, const float* __restrict__ W,
               unsigned short* __restrict__ Y) {
  __shared__ unsigned short At[128 * 40];  // [128][32] pad->40
  __shared__ unsigned short Wt[64 * 40];   // transposed: [n][k], [64][32] pad->40

  const int m0 = blockIdx.x * 128;
  const int n0 = blockIdx.y * 64;
  const int tid = threadIdx.x;
  const int wv = tid >> 5;
  const int lane = tid & 31;
  const int half = (lane >> 4) & 1;
  const int l16 = lane & 15;

  v8f acc[4];
#pragma unroll
  for (int t = 0; t < 4; ++t)
#pragma unroll
    for (int r = 0; r < 8; ++r) acc[t][r] = 0.f;

  for (int kk = 0; kk < DIM; kk += 32) {
    // stage A tile: 128x32 f32 -> bf16
    for (int i = tid; i < 1024; i += 256) {
      int r = i >> 3;
      int c4 = (i & 7) * 4;
      float4 f = *(const float4*)(X + (size_t)(m0 + r) * DIM + kk + c4);
      unsigned short* d = At + r * 40 + c4;
      d[0] = f32_to_bf16(f.x); d[1] = f32_to_bf16(f.y);
      d[2] = f32_to_bf16(f.z); d[3] = f32_to_bf16(f.w);
    }
    // stage W tile transposed: Wt[n][k] = W[kk+k][n0+n]
    for (int i = tid; i < 512; i += 256) {
      int kr = i >> 4;
      int c4 = (i & 15) * 4;
      float4 f = *(const float4*)(W + (size_t)(kk + kr) * DIM + n0 + c4);
      Wt[(c4 + 0) * 40 + kr] = f32_to_bf16(f.x);
      Wt[(c4 + 1) * 40 + kr] = f32_to_bf16(f.y);
      Wt[(c4 + 2) * 40 + kr] = f32_to_bf16(f.z);
      Wt[(c4 + 3) * 40 + kr] = f32_to_bf16(f.w);
    }
    __syncthreads();

    v16bf a = load_frag_A(At, 40, wv * 16 + l16, 0, half);
#pragma unroll
    for (int t = 0; t < 4; ++t) {
      v16bf b = load_frag_B(Wt, 40, t * 16 + l16, 0, half);
      acc[t] = wmma_bf16(a, b, acc[t]);
    }
    __syncthreads();
  }

  // epilogue: relu + bf16 store
#pragma unroll
  for (int t = 0; t < 4; ++t) {
#pragma unroll
    for (int r = 0; r < 8; ++r) {
      int m = m0 + wv * 16 + half * 8 + r;
      int n = n0 + t * 16 + l16;
      float v = acc[t][r];
      v = v > 0.f ? v : 0.f;
      Y[(size_t)m * DIM + n] = f32_to_bf16(v);
    }
  }
}

// ---------------------------------------------------------------------------
// Kernel 2: one-shot transposed bf16 convert of memory (the V operand).
//   mem:  [B][1024][512] f32   ->   vbT: [B][512][1024] bf16  (vbT[b][d][k])
// Tiled 64x64 through LDS so both global streams are coalesced.
// ---------------------------------------------------------------------------
__global__ void __launch_bounds__(256)
transpose_convert_bf16(const float* __restrict__ mem, unsigned short* __restrict__ vbT) {
  __shared__ unsigned short t[64 * 72];
  const int b = blockIdx.z;
  const int k0 = blockIdx.x * 64;   // key block
  const int d0 = blockIdx.y * 64;   // dim block
  const int tid = threadIdx.x;

  for (int i = tid; i < 1024; i += 256) {   // 64x64 floats = 1024 float4
    int r = i >> 4;
    int c4 = (i & 15) * 4;
    float4 f = *(const float4*)(mem + ((size_t)b * SEQ + k0 + r) * DIM + d0 + c4);
    unsigned short* d = t + r * 72 + c4;
    d[0] = f32_to_bf16(f.x); d[1] = f32_to_bf16(f.y);
    d[2] = f32_to_bf16(f.z); d[3] = f32_to_bf16(f.w);
  }
  __syncthreads();
  for (int i = tid; i < 512; i += 256) {    // 64x64 ushorts = 512 uint4
    int c = i >> 3;            // output row within dim block
    int r8 = (i & 7) * 8;
    union { uint4 q; unsigned short s[8]; } u;
#pragma unroll
    for (int j = 0; j < 8; ++j) u.s[j] = t[(r8 + j) * 72 + c];
    *(uint4*)(vbT + ((size_t)b * DIM + d0 + c) * SEQ + k0 + r8) = u.q;
  }
}

// ---------------------------------------------------------------------------
// Kernel 3: fused flash attention, software-pipelined async staging.
// Grid: (Sq/64, B). Block: 256 threads = 8 waves.
// Wave w owns output columns [64w, 64w+64) for all 64 queries of the block.
// Per-thread async issue pattern per tile: 16 K-copies then 16 V-copies, so
// "asynccnt <= 16" precisely drains everything older than the newest group.
// ---------------------------------------------------------------------------
#define XQ_STRIDE 520   // ushorts, 64 rows
#define KB_STRIDE 520   // ushorts, 64 rows
#define VT_STRIDE 72    // ushorts, 512 rows ([att_col][key] tile)
#define PS_STRIDE 68    // floats,  64 rows
#define PB_STRIDE 72    // ushorts, 64 rows

#define OFF_XQ 0
#define OFF_KB (OFF_XQ + 64 * XQ_STRIDE * 2)        // 66560
#define OFF_VT (OFF_KB + 64 * KB_STRIDE * 2)        // 133120
#define OFF_PS (OFF_VT + 512 * VT_STRIDE * 2)       // 206848
#define OFF_PB (OFF_PS + 64 * PS_STRIDE * 4)        // 224256
#define OFF_MR (OFF_PB + 64 * PB_STRIDE * 2)        // 233472
#define OFF_LR (OFF_MR + 64 * 4)
#define OFF_AR (OFF_LR + 64 * 4)
#define OFF_MSK (OFF_AR + 64 * 4)
#define OFF_RM (OFF_MSK + 64 * 4)                   // partial-max [64][4]
#define OFF_RL (OFF_RM + 256 * 4)                   // partial-sum [64][4]
#define ATTN_SMEM_BYTES (OFF_RL + 256 * 4)          // 236544

__global__ void __launch_bounds__(256)
attn_kernel(const unsigned short* __restrict__ xb,   // relu(inp@Wi)  bf16 [B*Sq,512]
            const unsigned short* __restrict__ mb,   // relu(mem@Wm)  bf16 [B*Sk,512]
            const unsigned short* __restrict__ vbT,  // memory^T      bf16 [B][512][1024]
            const int* __restrict__ mask,            // [B,Sk]
            float* __restrict__ out) {               // [B,Sq,1024]
  extern __shared__ __align__(16) char smem[];
  unsigned short* xq = (unsigned short*)(smem + OFF_XQ);
  unsigned short* kb = (unsigned short*)(smem + OFF_KB);
  unsigned short* vT = (unsigned short*)(smem + OFF_VT);
  float*          pS = (float*)(smem + OFF_PS);
  unsigned short* pB = (unsigned short*)(smem + OFF_PB);
  float*        mrow = (float*)(smem + OFF_MR);
  float*        lrow = (float*)(smem + OFF_LR);
  float*        arow = (float*)(smem + OFF_AR);
  int*         smask = (int*)(smem + OFF_MSK);
  float*        redm = (float*)(smem + OFF_RM);
  float*        redl = (float*)(smem + OFF_RL);

  const int b = blockIdx.y;
  const int q0 = blockIdx.x * 64;
  const int tid = threadIdx.x;
  const int wv = tid >> 5;
  const int lane = tid & 31;
  const int half = (lane >> 4) & 1;
  const int l16 = lane & 15;
  const float scale = 0.044194173824159216f;  // 1/sqrt(512)

  // ---- prologue: async-issue Q tile, K(0), V(0) ----
  for (int i = tid; i < 4096; i += 256) {    // Q: 16 issues/thread
    int r = i >> 6;
    int c8 = (i & 63) * 8;
    copy16B(xq + r * XQ_STRIDE + c8, xb + ((size_t)b * SEQ + q0 + r) * DIM + c8);
  }
  for (int i = tid; i < 4096; i += 256) {    // K(0): 16 issues/thread
    int r = i >> 6;
    int c8 = (i & 63) * 8;
    copy16B(kb + r * KB_STRIDE + c8, mb + ((size_t)b * SEQ + r) * DIM + c8);
  }
  for (int i = tid; i < 4096; i += 256) {    // V(0): 16 issues/thread
    int c = i >> 3;
    int j8 = (i & 7) * 8;
    copy16B(vT + c * VT_STRIDE + j8, vbT + ((size_t)b * DIM + c) * SEQ + j8);
  }
  if (tid < 64) { mrow[tid] = -3.0e38f; lrow[tid] = 0.f; }

  v8f acc[16];  // [qi][nj], wave-local context slice 64q x 64n
#pragma unroll
  for (int t = 0; t < 16; ++t)
#pragma unroll
    for (int r = 0; r < 8; ++r) acc[t][r] = 0.f;

  const int t0 = wv * 2;       // this wave's pair of S tiles
  const int qi = t0 >> 2;      // same q-subtile for both
  const int nb = (t0 & 3);     // first key-subtile

  // softmax lane mapping: 4 threads per query row
  const int srow = tid >> 2;
  const int sseg = tid & 3;

  for (int kt = 0; kt < 16; ++kt) {
    const int k0 = kt * 64;
    if (tid < 64) smask[tid] = mask[(size_t)b * SEQ + k0 + tid];

    // Q + K(kt) landed (newest <=16 outstanding are V(kt) copies)
    async_wait_le16();
    __syncthreads();

    // ---- S = xq @ kb^T  (two 16x16 tiles per wave, K = 512) ----
    v8f s0, s1;
#pragma unroll
    for (int r = 0; r < 8; ++r) { s0[r] = 0.f; s1[r] = 0.f; }
    for (int kk = 0; kk < DIM; kk += 32) {
      v16bf a  = load_frag_A(xq, XQ_STRIDE, qi * 16 + l16, kk, half);
      v16bf b0 = load_frag_B(kb, KB_STRIDE, nb * 16 + l16, kk, half);
      v16bf b1 = load_frag_B(kb, KB_STRIDE, (nb + 1) * 16 + l16, kk, half);
      s0 = wmma_bf16(a, b0, s0);
      s1 = wmma_bf16(a, b1, s1);
    }
    // scale + mask -> pS
#pragma unroll
    for (int r = 0; r < 8; ++r) {
      int m = qi * 16 + half * 8 + r;
      int n0c = nb * 16 + l16;
      int n1c = n0c + 16;
      float v0 = s0[r] * scale;
      float v1 = s1[r] * scale;
      if (!smask[n0c]) v0 = -1e30f;
      if (!smask[n1c]) v1 = -1e30f;
      pS[m * PS_STRIDE + n0c] = v0;
      pS[m * PS_STRIDE + n1c] = v1;
    }
    __syncthreads();   // all waves done reading kb

    // prefetch K(kt+1): overlaps softmax + P@V below
    if (kt < 15) {
      for (int i = tid; i < 4096; i += 256) {
        int r = i >> 6;
        int c8 = (i & 63) * 8;
        copy16B(kb + r * KB_STRIDE + c8,
                mb + ((size_t)b * SEQ + k0 + 64 + r) * DIM + c8);
      }
    }

    // ---- online softmax, 4 threads per row ----
    {
      float mx = -3.0e38f;
#pragma unroll
      for (int j = 0; j < 16; ++j)
        mx = fmaxf(mx, pS[srow * PS_STRIDE + sseg * 16 + j]);
      redm[srow * 4 + sseg] = mx;
    }
    __syncthreads();
    {
      float mold = mrow[srow];
      float mxn = fmaxf(fmaxf(fmaxf(redm[srow * 4 + 0], redm[srow * 4 + 1]),
                              fmaxf(redm[srow * 4 + 2], redm[srow * 4 + 3])), mold);
      float l = 0.f;
#pragma unroll
      for (int j = 0; j < 16; ++j) {
        float e = __expf(pS[srow * PS_STRIDE + sseg * 16 + j] - mxn);
        pB[srow * PB_STRIDE + sseg * 16 + j] = f32_to_bf16(e);
        l += e;
      }
      redl[srow * 4 + sseg] = l;
      __syncthreads();
      if (sseg == 0) {
        float lsum = redl[srow * 4 + 0] + redl[srow * 4 + 1] +
                     redl[srow * 4 + 2] + redl[srow * 4 + 3];
        float al = __expf(mold - mxn);
        lrow[srow] = lrow[srow] * al + lsum;
        mrow[srow] = mxn;
        arow[srow] = al;
      }
    }
    // V(kt) landed (outstanding, if any, are the K(kt+1) issues above)
    if (kt < 15) async_wait_le16(); else async_wait_0();
    __syncthreads();

    // ---- rescale accumulators + accumulate P @ V  (K = 64 keys) ----
    v16bf vb0[4], vb1[4];
#pragma unroll
    for (int nj = 0; nj < 4; ++nj) {
      int coln = wv * 64 + nj * 16 + l16;
      vb0[nj] = load_frag_B(vT, VT_STRIDE, coln, 0, half);
      vb1[nj] = load_frag_B(vT, VT_STRIDE, coln, 32, half);
    }
#pragma unroll
    for (int qi2 = 0; qi2 < 4; ++qi2) {
      float al[8];
#pragma unroll
      for (int r = 0; r < 8; ++r) al[r] = arow[qi2 * 16 + half * 8 + r];
      v16bf a0 = load_frag_A(pB, PB_STRIDE, qi2 * 16 + l16, 0, half);
      v16bf a1 = load_frag_A(pB, PB_STRIDE, qi2 * 16 + l16, 32, half);
#pragma unroll
      for (int nj = 0; nj < 4; ++nj) {
        v8f c = acc[qi2 * 4 + nj];
#pragma unroll
        for (int r = 0; r < 8; ++r) c[r] *= al[r];
        c = wmma_bf16(a0, vb0[nj], c);
        c = wmma_bf16(a1, vb1[nj], c);
        acc[qi2 * 4 + nj] = c;
      }
    }
    __syncthreads();   // all waves done reading vT

    // prefetch V(kt+1): overlaps next tile's S phase
    if (kt < 15) {
      for (int i = tid; i < 4096; i += 256) {
        int c = i >> 3;
        int j8 = (i & 7) * 8;
        copy16B(vT + c * VT_STRIDE + j8,
                vbT + ((size_t)b * DIM + c) * SEQ + k0 + 64 + j8);
      }
    }
  }

  // context = acc / l  -> left half of out
#pragma unroll
  for (int qi2 = 0; qi2 < 4; ++qi2) {
#pragma unroll
    for (int r = 0; r < 8; ++r) {
      int m = qi2 * 16 + half * 8 + r;
      float inv = 1.0f / lrow[m];
#pragma unroll
      for (int nj = 0; nj < 4; ++nj) {
        int col = wv * 64 + nj * 16 + l16;
        out[((size_t)b * SEQ + q0 + m) * 1024 + col] = acc[qi2 * 4 + nj][r] * inv;
      }
    }
  }
}

// ---------------------------------------------------------------------------
// Kernel 4: out[b,q,512:1024] = inputs[b,q,:]
// ---------------------------------------------------------------------------
__global__ void __launch_bounds__(256)
concat_inputs_kernel(const float* __restrict__ inp, float* __restrict__ out) {
  size_t i = ((size_t)blockIdx.x * blockDim.x + threadIdx.x) * 4;
  size_t row = i >> 9;        // / 512
  size_t c = i & 511;
  *(float4*)(out + row * 1024 + 512 + c) = *(const float4*)(inp + i);
}

// ---------------------------------------------------------------------------
extern "C" void kernel_launch(void* const* d_in, const int* in_sizes, int n_in,
                              void* d_out, int out_size, void* d_ws, size_t ws_size,
                              hipStream_t stream) {
  (void)in_sizes; (void)n_in; (void)out_size; (void)ws_size;
  const float* inputs = (const float*)d_in[0];   // [32,1024,512]
  const float* memory = (const float*)d_in[1];   // [32,1024,512]
  const int*   mmask  = (const int*)d_in[2];     // [32,1024]
  const float* Wi     = (const float*)d_in[3];   // [512,512]
  const float* Wm     = (const float*)d_in[4];   // [512,512]
  float* out = (float*)d_out;                    // [32,1024,1024]

  const size_t NE = (size_t)BATCH * SEQ * DIM;   // 16,777,216 elements
  unsigned short* xb  = (unsigned short*)d_ws;          // relu(inp@Wi) bf16
  unsigned short* mbp = xb + NE;                        // relu(mem@Wm) bf16
  unsigned short* vbT = mbp + NE;                       // memory^T bf16 [B][512][1024]

  // projections: M = 32*1024 = 32768 rows
  dim3 pgrid(32768 / 128, DIM / 64);
  proj_relu_bf16<<<pgrid, 256, 0, stream>>>(inputs, Wi, xb);
  proj_relu_bf16<<<pgrid, 256, 0, stream>>>(memory, Wm, mbp);

  // transposed V convert (one-shot)
  dim3 tgrid(SEQ / 64, DIM / 64, BATCH);
  transpose_convert_bf16<<<tgrid, 256, 0, stream>>>(memory, vbT);

  // fused attention (large dynamic LDS: ~231 KB of the 320 KB/WGP)
  (void)hipFuncSetAttribute((const void*)attn_kernel,
                            hipFuncAttributeMaxDynamicSharedMemorySize,
                            ATTN_SMEM_BYTES);
  dim3 agrid(SEQ / 64, BATCH);
  attn_kernel<<<agrid, 256, ATTN_SMEM_BYTES, stream>>>(xb, mbp, vbT, mmask, out);

  // concat right half
  concat_inputs_kernel<<<(unsigned)(NE / 4 / 256), 256, 0, stream>>>(inputs, out);
}
